// TransformerEncoderLayer_4398046511422
// MI455X (gfx1250) — matrix operand order repaired and checked by reading, not compiled, not run on hardware
//
#include <hip/hip_runtime.h>
#include <math.h>

// ---------------------------------------------------------------------------
// Transformer encoder layer for MI455X (gfx1250, wave32, WMMA bf16 path,
// async global->LDS staging). B=8, T=1024, D=1024, H=16, HD=64.
// ---------------------------------------------------------------------------

typedef __attribute__((ext_vector_type(16))) __bf16 v16bf;
typedef __attribute__((ext_vector_type(8)))  float  v8f;

#define Bdim 8
#define Tdim 1024
#define Ddim 1024
#define Hdim 16
#define HDdim 64
#define MTOK (Bdim * Tdim)   // 8192 tokens

enum { EPI_RES_F32 = 0, EPI_GELU_BF16 = 1, EPI_PACK_QKV = 2 };

__device__ __forceinline__ unsigned short f32_to_bf16(float f) {
  union { float f; unsigned u; } x; x.f = f;
  unsigned r = x.u + 0x7FFFu + ((x.u >> 16) & 1u);   // round-to-nearest-even
  return (unsigned short)(r >> 16);
}

__device__ __forceinline__ v8f wmma_bf16(v16bf a, v16bf b, v8f c) {
  return __builtin_amdgcn_wmma_f32_16x16x32_bf16(
      false, a, false, b, (short)0, c, false, false);
}

// Async direct global->LDS copy of 16 bytes per lane (ASYNCcnt-tracked).
// lds_off: low-32 bits of the generic LDS address (= DS offset).
// gofs:    per-lane byte offset added to the uniform 64-bit base (GVS mode).
__device__ __forceinline__ void async_copy_b128(unsigned lds_off, unsigned gofs,
                                                const void* base) {
  asm volatile("global_load_async_to_lds_b128 %0, %1, %2"
               :: "v"(lds_off), "v"(gofs), "s"(base) : "memory");
}
__device__ __forceinline__ void async_wait0() {
  asm volatile("s_wait_asynccnt 0x0" ::: "memory");
}

// A-fragment (16x32 bf16, ISA 7.12.2): lane L holds row M=L%16.
__device__ __forceinline__ v16bf load_a_frag(const unsigned short* base, int stride,
                                             int rbase, int koff) {
  int lane = threadIdx.x & 31;
  int hf = (lane >> 4) & 1;
  int m  = lane & 15;
  const unsigned short* rowp = base + (rbase + m) * stride;
  int k0 = koff + hf * 8;
  int k1 = koff + 16 + hf * 8;
  union { v16bf v; unsigned u[8]; } f;
#pragma unroll
  for (int p = 0; p < 4; ++p) {
    f.u[p]     = *(const unsigned*)(rowp + k0 + 2 * p);
    f.u[p + 4] = *(const unsigned*)(rowp + k1 + 2 * p);
  }
  return f.v;
}

// B-fragment (32x16 bf16) from an LDS tile stored transposed as [N][K].
__device__ __forceinline__ v16bf load_b_frag(const unsigned short* base, int stride,
                                             int nbase, int koff) {
  int lane = threadIdx.x & 31;
  int hf = (lane >> 4) & 1;
  int m  = lane & 15;
  const unsigned short* rowp = base + (nbase + m) * stride;
  int k0 = koff + hf * 8;
  int k1 = koff + 16 + hf * 8;
  union { v16bf v; unsigned u[8]; } f;
#pragma unroll
  for (int p = 0; p < 4; ++p) {
    f.u[p]     = *(const unsigned*)(rowp + k0 + 2 * p);
    f.u[p + 4] = *(const unsigned*)(rowp + k1 + 2 * p);
  }
  return f.v;
}

// ---------------------------------------------------------------------------
__global__ __launch_bounds__(256) void cvt_bf16_kernel(const float* __restrict__ src,
                                                       unsigned short* __restrict__ dst,
                                                       int n) {
  int i = blockIdx.x * 256 + threadIdx.x;
  if (i < n) dst[i] = f32_to_bf16(src[i]);
}

// ---------------------------------------------------------------------------
// bf16 WMMA GEMM: C[M,N] = A[M,K] * B[K,N] + bias, fused epilogues.
// grid = (N/128, M/128), block = 256 (8 waves); wave w owns rows [16w,16w+16)
// and all 128 columns (8 WMMAs per 32-wide k-step per wave).
// A tile staged via async global->LDS; B tile transposed manually.
// ---------------------------------------------------------------------------
__global__ __launch_bounds__(256) void gemm_bf16_kernel(
    const unsigned short* __restrict__ A, const unsigned short* __restrict__ B,
    const float* __restrict__ bias, const float* __restrict__ res,
    float* __restrict__ Cf, unsigned short* __restrict__ Cb,
    unsigned short* __restrict__ Qp, unsigned short* __restrict__ Kp,
    unsigned short* __restrict__ Vp,
    int M, int N, int K, int mode) {
  __shared__ __align__(16) unsigned short As[128 * 48];  // [row][k], stride 48
  __shared__ __align__(16) unsigned short Bs[128 * 48];  // transposed [n][k]

  int tid = threadIdx.x;
  int wave = tid >> 5, lane = tid & 31;
  int m15 = lane & 15, hf = (lane >> 4) & 1;
  int row0 = blockIdx.y * 128, col0 = blockIdx.x * 128;

  v8f acc[8];
#pragma unroll
  for (int n = 0; n < 8; ++n) acc[n] = (v8f){0,0,0,0,0,0,0,0};

  for (int k0 = 0; k0 < K; k0 += 32) {
    __syncthreads();
    {   // A tile 128x32: async copy, 32 bytes (2 x b128) per thread
      int r = tid >> 1, hseg = tid & 1;
      unsigned lds0 = (unsigned)(size_t)&As[r * 48 + hseg * 16];
      unsigned ga   = (unsigned)(((size_t)(row0 + r) * K + k0 + hseg * 16) * 2);
      async_copy_b128(lds0, ga, A);
      asm volatile("global_load_async_to_lds_b128 %0, %1, %2 offset:16"
                   :: "v"(lds0), "v"(ga), "s"(A) : "memory");
      if (k0 + 32 < K)
        __builtin_prefetch(A + (size_t)(row0 + r) * K + k0 + 32 + hseg * 16, 0, 3);
    }
    {   // B tile 32x128, transposed into Bs[n][k]
      int kk = tid >> 3, cseg = tid & 7;
      const uint4* bsrc = (const uint4*)(B + (size_t)(k0 + kk) * N + col0 + cseg * 16);
      uint4 b0 = bsrc[0], b1 = bsrc[1];
      union { uint4 q; unsigned short s[8]; } u0, u1; u0.q = b0; u1.q = b1;
#pragma unroll
      for (int j = 0; j < 8; ++j) {
        Bs[(cseg * 16 + j) * 48 + kk]     = u0.s[j];
        Bs[(cseg * 16 + 8 + j) * 48 + kk] = u1.s[j];
      }
      if (k0 + 32 < K)
        __builtin_prefetch(B + (size_t)(k0 + 32 + kk) * N + col0 + cseg * 16, 0, 3);
    }
    async_wait0();
    __syncthreads();

    v16bf a = load_a_frag(As, 48, wave * 16, 0);
#pragma unroll
    for (int n = 0; n < 8; ++n) {
      v16bf bfr = load_b_frag(Bs, 48, n * 16, 0);
      acc[n] = wmma_bf16(a, bfr, acc[n]);
    }
  }

  // epilogue (C layout: lane holds N = lane%16; VGPR r holds M = r + 8*half)
#pragma unroll
  for (int n = 0; n < 8; ++n) {
#pragma unroll
    for (int r = 0; r < 8; ++r) {
      int gr = row0 + wave * 16 + r + hf * 8;
      int gc = col0 + n * 16 + m15;
      float v = acc[n][r] + bias[gc];
      if (mode == EPI_RES_F32) {
        Cf[(size_t)gr * N + gc] = v + res[(size_t)gr * N + gc];
      } else if (mode == EPI_GELU_BF16) {
        float gv = 0.5f * v * (1.0f + erff(v * 0.70710678118f));
        Cb[(size_t)gr * N + gc] = f32_to_bf16(gv);
      } else {  // EPI_PACK_QKV
        int which = gc >> 10;
        int hh = (gc & 1023) >> 6;
        int dd = gc & 63;
        int bb = gr >> 10;
        int tt = gr & 1023;
        size_t dst = ((size_t)(bb * Hdim + hh) * Tdim + tt) * HDdim + dd;
        if (which == 0)      Qp[dst] = f32_to_bf16(v * 0.125f);  // 1/sqrt(64)
        else if (which == 1) Kp[dst] = f32_to_bf16(v);
        else                 Vp[dst] = f32_to_bf16(v);
      }
    }
  }
}

// ---------------------------------------------------------------------------
// Flash attention: grid=(T/64, B*H), block=128 (4 waves). Q/K tiles staged
// with async global->LDS; V staged transposed manually.
// ---------------------------------------------------------------------------
__global__ __launch_bounds__(128) void attn_kernel(
    const unsigned short* __restrict__ Qp, const unsigned short* __restrict__ Kp,
    const unsigned short* __restrict__ Vp, unsigned short* __restrict__ ctxb) {
  __shared__ __align__(16) unsigned short Qs[64 * 72];   // [q][d]
  __shared__ __align__(16) unsigned short Ks[64 * 72];   // [k][d]
  __shared__ __align__(16) unsigned short Vst[64 * 72];  // transposed [d][k]
  __shared__ __align__(16) unsigned short Ps[64 * 72];   // [q][k] probs

  int tid = threadIdx.x;
  int wave = tid >> 5, lane = tid & 31;
  int m15 = lane & 15, hf = (lane >> 4) & 1;
  int bh = blockIdx.y;
  int b = bh >> 4, h = bh & 15;
  int q0 = blockIdx.x * 64;
  const size_t headOff = (size_t)bh * Tdim * HDdim;

  // Q tile (pre-scaled by 1/8 at pack time): async copy
  for (int i = tid; i < 64 * 8; i += 128) {
    int r = i >> 3, seg = i & 7;
    unsigned lds = (unsigned)(size_t)&Qs[r * 72 + seg * 8];
    unsigned go  = (unsigned)((headOff + (size_t)(q0 + r) * HDdim + seg * 8) * 2);
    async_copy_b128(lds, go, Qp);
  }

  float mrow[8], lrow[8];
  v8f o[4];
#pragma unroll
  for (int n = 0; n < 4; ++n) o[n] = (v8f){0,0,0,0,0,0,0,0};
#pragma unroll
  for (int r = 0; r < 8; ++r) { mrow[r] = -1e30f; lrow[r] = 0.0f; }

  for (int j = 0; j < Tdim / 64; ++j) {
    __syncthreads();
    int k0 = j * 64;
    for (int i = tid; i < 64 * 8; i += 128) {
      int r = i >> 3, seg = i & 7;
      // K tile: async direct to LDS
      unsigned lds = (unsigned)(size_t)&Ks[r * 72 + seg * 8];
      unsigned go  = (unsigned)((headOff + (size_t)(k0 + r) * HDdim + seg * 8) * 2);
      async_copy_b128(lds, go, Kp);
      // V tile: manual transposed store
      uint4 vv = *(const uint4*)(Vp + headOff + (size_t)(k0 + r) * HDdim + seg * 8);
      union { uint4 q; unsigned short s[8]; } u; u.q = vv;
#pragma unroll
      for (int t2 = 0; t2 < 8; ++t2) Vst[(seg * 8 + t2) * 72 + r] = u.s[t2];
      if (j + 1 < Tdim / 64) {
        __builtin_prefetch(Kp + headOff + (size_t)(k0 + 64 + r) * HDdim + seg * 8, 0, 3);
        __builtin_prefetch(Vp + headOff + (size_t)(k0 + 64 + r) * HDdim + seg * 8, 0, 3);
      }
    }
    async_wait0();
    __syncthreads();

    // S = Q * K^T
    v8f s[4];
#pragma unroll
    for (int n = 0; n < 4; ++n) s[n] = (v8f){0,0,0,0,0,0,0,0};
#pragma unroll
    for (int kk = 0; kk < HDdim; kk += 32) {
      v16bf a = load_a_frag(Qs, 72, wave * 16, kk);
#pragma unroll
      for (int n = 0; n < 4; ++n) {
        v16bf bb = load_b_frag(Ks, 72, n * 16, kk);
        s[n] = wmma_bf16(a, bb, s[n]);
      }
    }

    // online softmax: row M = r + 8*half lives on one 16-lane group
#pragma unroll
    for (int r = 0; r < 8; ++r) {
      float vmax = fmaxf(fmaxf(s[0][r], s[1][r]), fmaxf(s[2][r], s[3][r]));
#pragma unroll
      for (int d2 = 1; d2 < 16; d2 <<= 1) vmax = fmaxf(vmax, __shfl_xor(vmax, d2, 32));
      float mn = fmaxf(mrow[r], vmax);
      float sc = __expf(mrow[r] - mn);
      mrow[r] = mn;
      float rs = 0.0f;
#pragma unroll
      for (int n = 0; n < 4; ++n) {
        float p = __expf(s[n][r] - mn);
        s[n][r] = p;
        rs += p;
      }
#pragma unroll
      for (int d2 = 1; d2 < 16; d2 <<= 1) rs += __shfl_xor(rs, d2, 32);
      lrow[r] = lrow[r] * sc + rs;
#pragma unroll
      for (int n = 0; n < 4; ++n) o[n][r] *= sc;
    }

    // spill P (bf16) to LDS; same-wave DS ops are in-order
#pragma unroll
    for (int n = 0; n < 4; ++n)
#pragma unroll
      for (int r = 0; r < 8; ++r)
        Ps[(wave * 16 + r + hf * 8) * 72 + n * 16 + m15] = f32_to_bf16(s[n][r]);

    // O += P * V
#pragma unroll
    for (int kk = 0; kk < 64; kk += 32) {
      v16bf a = load_a_frag(Ps, 72, wave * 16, kk);
#pragma unroll
      for (int n = 0; n < 4; ++n) {
        v16bf bb = load_b_frag(Vst, 72, n * 16, kk);
        o[n] = wmma_bf16(a, bb, o[n]);
      }
    }
  }

  // normalize and write ctx as bf16 [B*T, D]
#pragma unroll
  for (int r = 0; r < 8; ++r) {
    float inv = 1.0f / lrow[r];
    int tAbs = q0 + wave * 16 + r + hf * 8;
#pragma unroll
    for (int n = 0; n < 4; ++n) {
      int col = h * HDdim + n * 16 + m15;
      ctxb[((size_t)(b * Tdim + tAbs)) * Ddim + col] = f32_to_bf16(o[n][r] * inv);
    }
  }
}

// ---------------------------------------------------------------------------
// LayerNorm over last dim (D=1024); optional bf16 copy for the next GEMM.
// ---------------------------------------------------------------------------
__global__ __launch_bounds__(256) void ln_kernel(
    const float* __restrict__ x, const float* __restrict__ g,
    const float* __restrict__ bb, float* __restrict__ y,
    unsigned short* __restrict__ ybf, int D) {
  int row = blockIdx.x;
  const float* xr = x + (size_t)row * D;
  float s = 0.0f, s2 = 0.0f;
  for (int c = threadIdx.x; c < D; c += 256) {
    float v = xr[c]; s += v; s2 += v * v;
  }
#pragma unroll
  for (int d2 = 16; d2 >= 1; d2 >>= 1) {
    s  += __shfl_xor(s,  d2, 32);
    s2 += __shfl_xor(s2, d2, 32);
  }
  __shared__ float ps[8], ps2[8];
  int wave = threadIdx.x >> 5, lane = threadIdx.x & 31;
  if (lane == 0) { ps[wave] = s; ps2[wave] = s2; }
  __syncthreads();
  if (threadIdx.x == 0) {
    float t = 0.0f, t2 = 0.0f;
#pragma unroll
    for (int i = 0; i < 8; ++i) { t += ps[i]; t2 += ps2[i]; }
    ps[0] = t; ps2[0] = t2;
  }
  __syncthreads();
  float mean = ps[0] / (float)D;
  float var  = ps2[0] / (float)D - mean * mean;
  float inv  = rsqrtf(var + 1e-5f);
  for (int c = threadIdx.x; c < D; c += 256) {
    float v = (xr[c] - mean) * inv * g[c] + bb[c];
    y[(size_t)row * D + c] = v;
    if (ybf) ybf[(size_t)row * D + c] = f32_to_bf16(v);
  }
}

// ---------------------------------------------------------------------------
extern "C" void kernel_launch(void* const* d_in, const int* in_sizes, int n_in,
                              void* d_out, int out_size, void* d_ws, size_t ws_size,
                              hipStream_t stream) {
  (void)in_sizes; (void)n_in; (void)out_size; (void)ws_size;
  const float* X     = (const float*)d_in[0];
  const float* W_qkv = (const float*)d_in[1];
  const float* b_qkv = (const float*)d_in[2];
  const float* W_out = (const float*)d_in[3];
  const float* b_out = (const float*)d_in[4];
  const float* ln1_g = (const float*)d_in[5];
  const float* ln1_b = (const float*)d_in[6];
  const float* ln2_g = (const float*)d_in[7];
  const float* ln2_b = (const float*)d_in[8];
  const float* W_ff1 = (const float*)d_in[9];
  const float* b_ff1 = (const float*)d_in[10];
  const float* W_ff2 = (const float*)d_in[11];
  const float* b_ff2 = (const float*)d_in[12];
  float* out = (float*)d_out;

  char* w = (char*)d_ws;
  size_t off = 0;
  auto alloc = [&](size_t bytes) -> void* {
    void* p = w + off;
    off += (bytes + 255) & ~(size_t)255;
    return p;
  };
  unsigned short* Xb     = (unsigned short*)alloc((size_t)MTOK * Ddim * 2);
  unsigned short* Wqkvb  = (unsigned short*)alloc((size_t)Ddim * 3 * Ddim * 2);
  unsigned short* Woutb  = (unsigned short*)alloc((size_t)Ddim * Ddim * 2);
  unsigned short* Wff1b  = (unsigned short*)alloc((size_t)Ddim * 4 * Ddim * 2);
  unsigned short* Wff2b  = (unsigned short*)alloc((size_t)4 * Ddim * Ddim * 2);
  unsigned short* Qp     = (unsigned short*)alloc((size_t)Bdim * Hdim * Tdim * HDdim * 2);
  unsigned short* Kp     = (unsigned short*)alloc((size_t)Bdim * Hdim * Tdim * HDdim * 2);
  unsigned short* Vp     = (unsigned short*)alloc((size_t)Bdim * Hdim * Tdim * HDdim * 2);
  unsigned short* ctxb   = (unsigned short*)alloc((size_t)MTOK * Ddim * 2);
  float*          res1   = (float*)alloc((size_t)MTOK * Ddim * 4);
  float*          X1     = (float*)alloc((size_t)MTOK * Ddim * 4);
  unsigned short* X1b    = (unsigned short*)alloc((size_t)MTOK * Ddim * 2);
  unsigned short* H1b    = (unsigned short*)alloc((size_t)MTOK * 4 * Ddim * 2);
  float*          res2   = (float*)alloc((size_t)MTOK * Ddim * 4);

  auto cvt = [&](const float* s, unsigned short* d, int n) {
    cvt_bf16_kernel<<<(n + 255) / 256, 256, 0, stream>>>(s, d, n);
  };
  cvt(X,     Xb,    MTOK * Ddim);
  cvt(W_qkv, Wqkvb, Ddim * 3 * Ddim);
  cvt(W_out, Woutb, Ddim * Ddim);
  cvt(W_ff1, Wff1b, Ddim * 4 * Ddim);
  cvt(W_ff2, Wff2b, 4 * Ddim * Ddim);

  // QKV projection, packed per-head (Q pre-scaled by 1/sqrt(HD))
  gemm_bf16_kernel<<<dim3(3 * Ddim / 128, MTOK / 128), 256, 0, stream>>>(
      Xb, Wqkvb, b_qkv, nullptr, nullptr, nullptr, Qp, Kp, Vp,
      MTOK, 3 * Ddim, Ddim, EPI_PACK_QKV);

  // attention -> ctx (bf16)
  attn_kernel<<<dim3(Tdim / 64, Bdim * Hdim), 128, 0, stream>>>(Qp, Kp, Vp, ctxb);

  // out projection + residual (X)
  gemm_bf16_kernel<<<dim3(Ddim / 128, MTOK / 128), 256, 0, stream>>>(
      ctxb, Woutb, b_out, X, res1, nullptr, nullptr, nullptr, nullptr,
      MTOK, Ddim, Ddim, EPI_RES_F32);

  // LN1 -> X1 (f32) + X1b (bf16)
  ln_kernel<<<MTOK, 256, 0, stream>>>(res1, ln1_g, ln1_b, X1, X1b, Ddim);

  // FF1 + GELU -> bf16
  gemm_bf16_kernel<<<dim3(4 * Ddim / 128, MTOK / 128), 256, 0, stream>>>(
      X1b, Wff1b, b_ff1, nullptr, nullptr, H1b, nullptr, nullptr, nullptr,
      MTOK, 4 * Ddim, Ddim, EPI_GELU_BF16);

  // FF2 + residual (X1)
  gemm_bf16_kernel<<<dim3(Ddim / 128, MTOK / 128), 256, 0, stream>>>(
      H1b, Wff2b, b_ff2, X1, res2, nullptr, nullptr, nullptr, nullptr,
      MTOK, Ddim, 4 * Ddim, EPI_RES_F32);

  // LN2 -> output (f32)
  ln_kernel<<<MTOK, 256, 0, stream>>>(res2, ln2_g, ln2_b, out, nullptr, Ddim);
}
